// MultiHeadAttention_88716844467031
// MI455X (gfx1250) — matrix operand order repaired and checked by reading, not compiled, not run on hardware
//
#include <hip/hip_runtime.h>
#include <hip/hip_bf16.h>

// ---- problem constants (from reference) ----
constexpr int cH   = 16;
constexpr int cNQ  = 2048;
constexpr int cNK  = 2048;
constexpr int cM   = 64;         // memory slots
constexpr int cNKM = cNK + cM;   // 2112
constexpr int cD   = 1024;
constexpr int cDK  = 64;
constexpr int cDV  = 64;
#define LN_EPS 1e-5f

// ---- CDNA5 WMMA types ----
typedef __bf16 bf16_t;
typedef __attribute__((ext_vector_type(16))) bf16_t v16bf;
typedef __attribute__((ext_vector_type(8)))  float  v8f;
typedef __attribute__((ext_vector_type(4)))  float  f32x4;
typedef __attribute__((ext_vector_type(4)))  unsigned int u32x4;
typedef __attribute__((ext_vector_type(2)))  unsigned int u32x2;

static __device__ __forceinline__ unsigned short f2bf(float f) {
  union { float f; unsigned int u; } c; c.f = f;
  unsigned int u = c.u;
  u += 0x7FFFu + ((u >> 16) & 1u);   // round-to-nearest-even
  return (unsigned short)(u >> 16);
}

// error-compensated split: x ~= hi + lo (both bf16), lo ~ 2^-8 of hi
static __device__ __forceinline__ void split_bf(float f, unsigned short& h, unsigned short& l) {
  h = f2bf(f);
  union { unsigned int u; float fv; } c; c.u = (unsigned int)h << 16;
  l = f2bf(f - c.fv);
}

// LDS tile row stride (bf16 elems): 40 elems = 80B = 16B-aligned, conflict-free
#define LDS_STRIDE 40

// load 4 consecutive f32, split, pack, 8B-store into hi/lo LDS tiles
static __device__ __forceinline__ void split_store4(
    const float* __restrict__ src, unsigned short* hi, unsigned short* lo, int off) {
  f32x4 x = *(const f32x4*)src;
  unsigned short h[4], l[4];
#pragma unroll
  for (int e = 0; e < 4; ++e) split_bf(x[e], h[e], l[e]);
  u32x2 hp, lp;
  hp[0] = (unsigned)h[0] | ((unsigned)h[1] << 16);
  hp[1] = (unsigned)h[2] | ((unsigned)h[3] << 16);
  lp[0] = (unsigned)l[0] | ((unsigned)l[1] << 16);
  lp[1] = (unsigned)l[2] | ((unsigned)l[3] << 16);
  *(u32x2*)(hi + off) = hp;
  *(u32x2*)(lo + off) = lp;
}

// A fragment: 16x32 bf16, rows striped in-lane per ISA table.
static __device__ __forceinline__ v16bf frag_a(const unsigned short* lds, int row0, int lane) {
  int r  = row0 + (lane & 15);
  int kb = (lane & 16) ? 8 : 0;
  union { v16bf v; u32x4 q[2]; } f;
  f.q[0] = *(const u32x4*)(lds + r * LDS_STRIDE + kb);
  f.q[1] = *(const u32x4*)(lds + r * LDS_STRIDE + kb + 16);
  return f.v;
}

// B fragment: 32x16 bf16, LDS stored [n][k]; K striped per ISA table (matches frag_a).
static __device__ __forceinline__ v16bf frag_b(const unsigned short* lds, int col0, int lane) {
  int n  = col0 + (lane & 15);
  int kb = (lane & 16) ? 8 : 0;
  union { v16bf v; u32x4 q[2]; } f;
  f.q[0] = *(const u32x4*)(lds + n * LDS_STRIDE + kb);
  f.q[1] = *(const u32x4*)(lds + n * LDS_STRIDE + kb + 16);
  return f.v;
}

static __device__ __forceinline__ v8f wmma_bf(v16bf a, v16bf b, v8f c) {
  return __builtin_amdgcn_wmma_f32_16x16x32_bf16(false, a, false, b, (short)0, c, false, false);
}

// bf16x3: c += a_lo*b_hi + a_hi*b_lo + a_hi*b_hi  (small terms first)
static __device__ __forceinline__ v8f wmma3(v16bf ah, v16bf al, v16bf bh, v16bf bl, v8f c) {
  c = wmma_bf(al, bh, c);
  c = wmma_bf(ah, bl, c);
  c = wmma_bf(ah, bh, c);
  return c;
}

// =====================================================================
// Kernel 1: Y = X(MxK) * W(NxK)^T + bias.  headed=1 -> Y[h][m][64] layout
// WG tile 128x64, 8 waves each 32x32 (2x2 tiles, bf16x3).  K % 32 == 0.
// =====================================================================
__global__ __launch_bounds__(256) void k_gemm_bias(
    const float* __restrict__ X, const float* __restrict__ W,
    const float* __restrict__ bias, float* __restrict__ Y,
    int M, int N, int K, int headed) {
  __shared__ unsigned short lAh[128 * LDS_STRIDE];
  __shared__ unsigned short lAl[128 * LDS_STRIDE];
  __shared__ unsigned short lBh[64 * LDS_STRIDE];
  __shared__ unsigned short lBl[64 * LDS_STRIDE];
  const int tid = threadIdx.x, lane = tid & 31, wid = tid >> 5;
  const int wm = wid & 3, wn = wid >> 2;
  const int m0 = blockIdx.y * 128, n0 = blockIdx.x * 64;

  v8f acc[2][2];
  const v8f vz = {0.f,0.f,0.f,0.f,0.f,0.f,0.f,0.f};
  acc[0][0] = vz; acc[0][1] = vz; acc[1][0] = vz; acc[1][1] = vz;

  for (int kk = 0; kk < K; kk += 32) {
    __syncthreads();
    if (kk + 32 < K)  // prefetch next A chunk -> global_prefetch_b8
      __builtin_prefetch(X + (size_t)(m0 + (tid >> 1)) * K + kk + 32, 0, 2);
    for (int i = tid * 4; i < 128 * 32; i += 1024) {
      int r = i >> 5, c = i & 31;
      split_store4(X + (size_t)(m0 + r) * K + kk + c, lAh, lAl, r * LDS_STRIDE + c);
    }
    for (int i = tid * 4; i < 64 * 32; i += 1024) {
      int r = i >> 5, c = i & 31;
      split_store4(W + (size_t)(n0 + r) * K + kk + c, lBh, lBl, r * LDS_STRIDE + c);
    }
    __syncthreads();
    v16bf ah[2], al[2], bh[2], bl[2];
#pragma unroll
    for (int t = 0; t < 2; ++t) {
      ah[t] = frag_a(lAh, wm * 32 + t * 16, lane);
      al[t] = frag_a(lAl, wm * 32 + t * 16, lane);
      bh[t] = frag_b(lBh, wn * 32 + t * 16, lane);
      bl[t] = frag_b(lBl, wn * 32 + t * 16, lane);
    }
#pragma unroll
    for (int tm = 0; tm < 2; ++tm)
#pragma unroll
      for (int tn = 0; tn < 2; ++tn)
        acc[tm][tn] = wmma3(ah[tm], al[tm], bh[tn], bl[tn], acc[tm][tn]);
  }

  const int lm = (lane & 16) ? 8 : 0, ln = lane & 15;
#pragma unroll
  for (int tm = 0; tm < 2; ++tm)
#pragma unroll
    for (int tn = 0; tn < 2; ++tn)
#pragma unroll
      for (int r = 0; r < 8; ++r) {
        int m = m0 + wm * 32 + tm * 16 + lm + r;
        int n = n0 + wn * 32 + tn * 16 + ln;
        float v = acc[tm][tn][r] + bias[n];
        if (headed) Y[((size_t)(n >> 6) * M + m) * 64 + (n & 63)] = v;
        else        Y[(size_t)m * N + n] = v;
      }
}

// =====================================================================
// Kernel 2: scores + reweight + bias + mask -> r_att[h][q][2112]
// WG tile 128x64 per head.  Block x==32 is entirely memory-slot columns
// (from m_k[h][dk][ms]), blocks 0..31 entirely regular -> uniform branches.
// =====================================================================
__global__ __launch_bounds__(256) void k_scores(
    const float* __restrict__ qbuf, const float* __restrict__ kbuf,
    const float* __restrict__ m_k, const float* __restrict__ aw,
    const float* __restrict__ pa, const unsigned char* __restrict__ msk,
    float* __restrict__ r_att) {
  __shared__ unsigned short lAh[128 * LDS_STRIDE];
  __shared__ unsigned short lAl[128 * LDS_STRIDE];
  __shared__ unsigned short lBh[64 * LDS_STRIDE];
  __shared__ unsigned short lBl[64 * LDS_STRIDE];
  const int tid = threadIdx.x, lane = tid & 31, wid = tid >> 5;
  const int wm = wid & 3, wn = wid >> 2;
  const int m0 = blockIdx.y * 128, n0 = blockIdx.x * 64, h = blockIdx.z;
  const bool memblk = (n0 >= cNK);   // uniform across the block

  v8f acc[2][2];
  const v8f vz = {0.f,0.f,0.f,0.f,0.f,0.f,0.f,0.f};
  acc[0][0] = vz; acc[0][1] = vz; acc[1][0] = vz; acc[1][1] = vz;

  for (int kk = 0; kk < cDK; kk += 32) {
    __syncthreads();
    for (int i = tid * 4; i < 128 * 32; i += 1024) {
      int r = i >> 5, c = i & 31;
      split_store4(qbuf + ((size_t)h * cNQ + m0 + r) * cDK + kk + c,
                   lAh, lAl, r * LDS_STRIDE + c);
    }
    if (!memblk) {
      for (int i = tid * 4; i < 64 * 32; i += 1024) {
        int r = i >> 5, c = i & 31;
        split_store4(kbuf + ((size_t)h * cNK + n0 + r) * cDK + kk + c,
                     lBh, lBl, r * LDS_STRIDE + c);
      }
    } else {
      for (int i = tid; i < 64 * 32; i += 256) {
        int r = i >> 5, c = i & 31;   // r = memory slot, c = dk offset
        float kv = m_k[((size_t)h * cDK + kk + c) * cM + r];  // (1,H,DK,M)
        unsigned short hh, ll; split_bf(kv, hh, ll);
        lBh[r * LDS_STRIDE + c] = hh;
        lBl[r * LDS_STRIDE + c] = ll;
      }
    }
    __syncthreads();
    v16bf ah[2], al[2], bh[2], bl[2];
#pragma unroll
    for (int t = 0; t < 2; ++t) {
      ah[t] = frag_a(lAh, wm * 32 + t * 16, lane);
      al[t] = frag_a(lAl, wm * 32 + t * 16, lane);
      bh[t] = frag_b(lBh, wn * 32 + t * 16, lane);
      bl[t] = frag_b(lBl, wn * 32 + t * 16, lane);
    }
#pragma unroll
    for (int tm = 0; tm < 2; ++tm)
#pragma unroll
      for (int tn = 0; tn < 2; ++tn)
        acc[tm][tn] = wmma3(ah[tm], al[tm], bh[tn], bl[tn], acc[tm][tn]);
  }

  const int lm = (lane & 16) ? 8 : 0, ln = lane & 15;
#pragma unroll
  for (int tm = 0; tm < 2; ++tm)
#pragma unroll
    for (int tn = 0; tn < 2; ++tn)
#pragma unroll
      for (int r = 0; r < 8; ++r) {
        int m = m0 + wm * 32 + tm * 16 + lm + r;
        int j = n0 + wn * 32 + tn * 16 + ln;
        float s = acc[tm][tn][r] * 0.125f;   // 1/sqrt(DK)
        float ra;
        if (!memblk) {
          size_t ix = (size_t)m * cNK + j;
          float att = s * aw[ix] + pa[ix];
          ra = (msk[ix] != 0) ? -__builtin_inff() : att;
        } else {
          ra = s;                            // aw=1, pa=0, mask=false
        }
        r_att[((size_t)h * cNQ + m) * cNKM + j] = ra;
      }
}

// =====================================================================
// Kernel 3: per-row softmax stats over 2112 cols -> rowmax, 1/rowsum
// =====================================================================
__global__ __launch_bounds__(256) void k_rowstats(
    const float* __restrict__ r_att, float* __restrict__ rmax, float* __restrict__ rinv) {
  __shared__ float red[256];
  const int row = blockIdx.x, tid = threadIdx.x;
  const float* p = r_att + (size_t)row * cNKM;
  float mx = -__builtin_inff();
  for (int i = tid; i < cNKM; i += 256) mx = fmaxf(mx, p[i]);
  red[tid] = mx; __syncthreads();
  for (int s = 128; s > 0; s >>= 1) { if (tid < s) red[tid] = fmaxf(red[tid], red[tid + s]); __syncthreads(); }
  mx = red[0]; __syncthreads();
  float sm = 0.f;
  for (int i = tid; i < cNKM; i += 256) sm += __expf(p[i] - mx);
  red[tid] = sm; __syncthreads();
  for (int s = 128; s > 0; s >>= 1) { if (tid < s) red[tid] += red[tid + s]; __syncthreads(); }
  if (tid == 0) { rmax[row] = mx; rinv[row] = 1.f / red[0]; }
}

// =====================================================================
// Kernel 4: ctx[q][h*64+dv] = softmax(r_att) * V.
// p computed on the fly while staging A; V staged transposed [dv][j];
// rows >= NK (last two k-chunks, uniform) come from m_v[h][ms][dv].
// =====================================================================
__global__ __launch_bounds__(256) void k_pv(
    const float* __restrict__ r_att, const float* __restrict__ rmax,
    const float* __restrict__ rinv, const float* __restrict__ vbuf,
    const float* __restrict__ m_v, float* __restrict__ ctx) {
  __shared__ unsigned short lAh[128 * LDS_STRIDE];
  __shared__ unsigned short lAl[128 * LDS_STRIDE];
  __shared__ unsigned short lBh[64 * LDS_STRIDE];
  __shared__ unsigned short lBl[64 * LDS_STRIDE];
  const int tid = threadIdx.x, lane = tid & 31, wid = tid >> 5;
  const int wm = wid & 3, wn = wid >> 2;
  const int m0 = blockIdx.x * 128, h = blockIdx.y;

  v8f acc[2][2];
  const v8f vz = {0.f,0.f,0.f,0.f,0.f,0.f,0.f,0.f};
  acc[0][0] = vz; acc[0][1] = vz; acc[1][0] = vz; acc[1][1] = vz;

  for (int kk = 0; kk < cNKM; kk += 32) {
    const bool memrow = (kk >= cNK);   // uniform per iteration
    __syncthreads();
    for (int i = tid * 4; i < 128 * 32; i += 1024) {
      int r = i >> 5, c = i & 31;
      size_t rowix = (size_t)h * cNQ + (m0 + r);
      f32x4 x = *(const f32x4*)(r_att + rowix * cNKM + kk + c);
      float mxv = rmax[rowix], inv = rinv[rowix];
      unsigned short hh[4], ll[4];
#pragma unroll
      for (int e = 0; e < 4; ++e) {
        float pv = __expf(x[e] - mxv) * inv;   // exp(-inf)=0 for masked
        split_bf(pv, hh[e], ll[e]);
      }
      u32x2 hp, lp;
      hp[0] = (unsigned)hh[0] | ((unsigned)hh[1] << 16);
      hp[1] = (unsigned)hh[2] | ((unsigned)hh[3] << 16);
      lp[0] = (unsigned)ll[0] | ((unsigned)ll[1] << 16);
      lp[1] = (unsigned)ll[2] | ((unsigned)ll[3] << 16);
      *(u32x2*)(lAh + r * LDS_STRIDE + c) = hp;
      *(u32x2*)(lAl + r * LDS_STRIDE + c) = lp;
    }
    for (int i = tid * 4; i < 64 * 32; i += 1024) {
      int dv = i & 63, jl = i >> 6;
      int gj = kk + jl;
      const float* src = memrow
          ? (m_v + ((size_t)h * cM + (gj - cNK)) * cDV + dv)     // (1,H,M,DV)
          : (vbuf + ((size_t)h * cNK + gj) * cDV + dv);
      f32x4 x = *(const f32x4*)src;
#pragma unroll
      for (int e = 0; e < 4; ++e) {
        unsigned short hh, ll; split_bf(x[e], hh, ll);
        lBh[(dv + e) * LDS_STRIDE + jl] = hh;   // transposed [dv][j]
        lBl[(dv + e) * LDS_STRIDE + jl] = ll;
      }
    }
    __syncthreads();
    v16bf ah[2], al[2], bh[2], bl[2];
#pragma unroll
    for (int t = 0; t < 2; ++t) {
      ah[t] = frag_a(lAh, wm * 32 + t * 16, lane);
      al[t] = frag_a(lAl, wm * 32 + t * 16, lane);
      bh[t] = frag_b(lBh, wn * 32 + t * 16, lane);
      bl[t] = frag_b(lBl, wn * 32 + t * 16, lane);
    }
#pragma unroll
    for (int tm = 0; tm < 2; ++tm)
#pragma unroll
      for (int tn = 0; tn < 2; ++tn)
        acc[tm][tn] = wmma3(ah[tm], al[tm], bh[tn], bl[tn], acc[tm][tn]);
  }

  const int lm = (lane & 16) ? 8 : 0, ln = lane & 15;
#pragma unroll
  for (int tm = 0; tm < 2; ++tm)
#pragma unroll
    for (int tn = 0; tn < 2; ++tn)
#pragma unroll
      for (int r = 0; r < 8; ++r) {
        int m  = m0 + wm * 32 + tm * 16 + lm + r;
        int dv = wn * 32 + tn * 16 + ln;
        ctx[(size_t)m * cD + h * cDV + dv] = acc[tm][tn][r];
      }
}

// =====================================================================
// Kernel 5: out = LayerNorm(queries + obuf) * gamma + beta
// =====================================================================
__global__ __launch_bounds__(256) void k_layernorm(
    const float* __restrict__ q, const float* __restrict__ o,
    const float* __restrict__ gamma, const float* __restrict__ beta,
    float* __restrict__ out) {
  __shared__ float red[256];
  const int row = blockIdx.x, tid = threadIdx.x;
  const float* qr = q + (size_t)row * cD;
  const float* orow = o + (size_t)row * cD;
  float s = 0.f;
  for (int i = tid; i < cD; i += 256) s += qr[i] + orow[i];
  red[tid] = s; __syncthreads();
  for (int st = 128; st > 0; st >>= 1) { if (tid < st) red[tid] += red[tid + st]; __syncthreads(); }
  float mu = red[0] * (1.f / cD); __syncthreads();
  float v = 0.f;
  for (int i = tid; i < cD; i += 256) { float d = qr[i] + orow[i] - mu; v += d * d; }
  red[tid] = v; __syncthreads();
  for (int st = 128; st > 0; st >>= 1) { if (tid < st) red[tid] += red[tid + st]; __syncthreads(); }
  float rs = rsqrtf(red[0] * (1.f / cD) + LN_EPS);
  for (int i = tid; i < cD; i += 256)
    out[(size_t)row * cD + i] = (qr[i] + orow[i] - mu) * rs * gamma[i] + beta[i];
}

// =====================================================================
extern "C" void kernel_launch(void* const* d_in, const int* in_sizes, int n_in,
                              void* d_out, int out_size, void* d_ws, size_t ws_size,
                              hipStream_t stream) {
  (void)in_sizes; (void)n_in; (void)out_size; (void)ws_size;
  const float* queries = (const float*)d_in[0];
  const float* keys    = (const float*)d_in[1];
  const float* values  = (const float*)d_in[2];
  const float* aw      = (const float*)d_in[3];
  const float* pa      = (const float*)d_in[4];
  const unsigned char* msk = (const unsigned char*)d_in[5];  // jnp bool -> 1 byte
  const float* Wq = (const float*)d_in[6];  const float* bq = (const float*)d_in[7];
  const float* Wk = (const float*)d_in[8];  const float* bk = (const float*)d_in[9];
  const float* Wv = (const float*)d_in[10]; const float* bv = (const float*)d_in[11];
  const float* Wo = (const float*)d_in[12]; const float* bo = (const float*)d_in[13];
  const float* m_k = (const float*)d_in[14];
  const float* m_v = (const float*)d_in[15];
  const float* gamma = (const float*)d_in[16];
  const float* beta  = (const float*)d_in[17];

  float* out   = (float*)d_out;                    // (1, NQ, D)
  float* r_att = out + (size_t)cNQ * cD;           // (1, H, NQ, NK+M)

  float* ws   = (float*)d_ws;                      // ~42 MB total
  float* qbuf = ws;  ws += (size_t)cH * cNQ * cDK; // head-major Q
  float* kbuf = ws;  ws += (size_t)cH * cNK * cDK; // head-major K ([j][dk])
  float* vbuf = ws;  ws += (size_t)cH * cNK * cDV; // head-major V
  float* ctx  = ws;  ws += (size_t)cNQ * cD;       // attention context
  float* obuf = ws;  ws += (size_t)cNQ * cD;       // output projection
  float* rmax = ws;  ws += (size_t)cH * cNQ;
  float* rinv = ws;

  dim3 blk(256);
  dim3 gProj(cD / 64, cNQ / 128);
  k_gemm_bias<<<gProj, blk, 0, stream>>>(queries, Wq, bq, qbuf, cNQ, cD, cD, 1);
  k_gemm_bias<<<gProj, blk, 0, stream>>>(keys,    Wk, bk, kbuf, cNK, cD, cD, 1);
  k_gemm_bias<<<gProj, blk, 0, stream>>>(values,  Wv, bv, vbuf, cNK, cD, cD, 1);

  dim3 gS(cNKM / 64, cNQ / 128, cH);
  k_scores<<<gS, blk, 0, stream>>>(qbuf, kbuf, m_k, aw, pa, msk, r_att);

  k_rowstats<<<dim3(cH * cNQ), blk, 0, stream>>>(r_att, rmax, rinv);

  k_pv<<<dim3(cNQ / 128, cH), blk, 0, stream>>>(r_att, rmax, rinv, vbuf, m_v, ctx);

  k_gemm_bias<<<gProj, blk, 0, stream>>>(ctx, Wo, bo, obuf, cNQ, cD, cD, 0);

  k_layernorm<<<dim3(cNQ), blk, 0, stream>>>(queries, obuf, gamma, beta, out);
}
